// BiLSTMLayer_48112223650330
// MI455X (gfx1250) — matrix operand order repaired and checked by reading, not compiled, not run on hardware
//
#include <hip/hip_runtime.h>
#include <hip/hip_bf16.h>

typedef __attribute__((ext_vector_type(16))) __bf16 v16bf;
typedef __attribute__((ext_vector_type(8)))  __bf16 bf16x8;
typedef __attribute__((ext_vector_type(8)))  float  v8f;
typedef unsigned int v4u __attribute__((ext_vector_type(4)));
typedef int          v4i __attribute__((ext_vector_type(4)));
typedef int          v8i __attribute__((ext_vector_type(8)));

#define TT 512
#define BB 32
#define DD 512
#define HH 512
#define JC 16            // hidden columns owned per workgroup
#define WGD 32           // workgroups per direction (HH / JC)
#define NG 64            // gate columns per WG (4 * JC)
#define KT 1024          // concat K = D + H
#define AXSTR 520        // LDS row stride (halfs) for x half of A, padded
#define AHSTR 520        // LDS row stride (halfs) for h half of A, padded (TDM pad)
#define WSTR 1032        // LDS row stride (halfs) for W, padded
#define GSTR 68          // LDS row stride (floats) for gate exchange

__device__ __forceinline__ unsigned short f2bf(float f){
  unsigned u = __builtin_bit_cast(unsigned, f);
  unsigned r = 0x7FFFu + ((u >> 16) & 1u);   // round-to-nearest-even
  return (unsigned short)((u + r) >> 16);
}
// fast sigmoid / tanh using v_exp_f32 + v_rcp_f32 (keeps the serial per-step
// critical path short; avoids v_div_scale IEEE division chains)
__device__ __forceinline__ float sigmoidf_(float x){
  float e = __expf(-x);
  return __builtin_amdgcn_rcpf(1.0f + e);
}
__device__ __forceinline__ float tanhf_(float x){
  float xc = fminf(fmaxf(x, -15.0f), 15.0f);
  float e  = __expf(2.0f * xc);
  return 1.0f - 2.0f * __builtin_amdgcn_rcpf(e + 1.0f);
}

__global__ void __launch_bounds__(256, 1)
bilstm_init(unsigned* ws_words, int nwords){
  int i = blockIdx.x * blockDim.x + threadIdx.x;
  if (i < nwords) ws_words[i] = 0u;
}

// Persistent bidirectional LSTM. Grid = 64 WGs: [0,32) forward, [32,64) backward.
// Each WG owns JC=16 hidden columns; W_ih|W_hh slice (64 rows x 1024 K, bf16)
// lives in LDS for the whole sequence. Per timestep: TDM async-loads h into LDS,
// VALU stages x_t (fp32->bf16), then one fused [32x1024]x[1024x64] bf16 WMMA
// GEMM, gate nonlinearities, state update, bf16 h broadcast via L2, and a
// device-wide barrier.
__global__ void __launch_bounds__(256, 1)
bilstm_persistent(const float* __restrict__ x,
                  const int*   __restrict__ lengths,
                  const float* __restrict__ Wih_f, const float* __restrict__ Whh_f,
                  const float* __restrict__ bih_f, const float* __restrict__ bhh_f,
                  const float* __restrict__ Wih_b, const float* __restrict__ Whh_b,
                  const float* __restrict__ bih_b, const float* __restrict__ bhh_b,
                  float* __restrict__ out,
                  unsigned short* __restrict__ hbuf_f,
                  unsigned short* __restrict__ hbuf_b,
                  unsigned* __restrict__ bar)
{
  __shared__ unsigned short sW [NG * WSTR];   // 132096 B : bf16 weight slice [64][1024+pad]
  __shared__ unsigned short sAx[BB * AXSTR];  //  33280 B : bf16 x_t   [32][512+pad]
  __shared__ unsigned short sAh[BB * AHSTR];  //  33280 B : bf16 h     [32][512+pad] (TDM dest)
  __shared__ float          sG [BB * GSTR];   //   8704 B : fp32 gate tile exchange
  __shared__ float          sBias[NG];
  __shared__ int            sLen[BB];

  const int tid = threadIdx.x;
  const int dir = (blockIdx.x >= WGD) ? 1 : 0;
  const int wg  = blockIdx.x & (WGD - 1);
  const int j0  = wg * JC;

  const float* Wih = dir ? Wih_b : Wih_f;
  const float* Whh = dir ? Whh_b : Whh_f;
  const float* bih = dir ? bih_b : bih_f;
  const float* bhh = dir ? bhh_b : bhh_f;
  unsigned short* hbuf = dir ? hbuf_b : hbuf_f;
  unsigned* barp = bar + dir;

  // ---- preload: weight slice (fp32 -> bf16), fused biases, lengths ----
  for (int idx = tid; idx < NG * KT; idx += 256){
    int r = idx >> 10, k = idx & (KT - 1);
    int R = ((r >> 4) * HH) + j0 + (r & 15);       // gate-major row in [4H, .]
    float v = (k < DD) ? Wih[(size_t)R * DD + k] : Whh[(size_t)R * HH + (k - DD)];
    sW[r * WSTR + k] = f2bf(v);
  }
  if (tid < NG){
    int R = ((tid >> 4) * HH) + j0 + (tid & 15);
    sBias[tid] = bih[R] + bhh[R];
  }
  if (tid < BB) sLen[tid] = lengths[tid];
  __syncthreads();

  // ---- TDM descriptor: 2D tile [rows=32][cols=512] bf16, global hbuf -> sAh,
  //      LDS padding 4 DWORDs every 256 DWORDs => padded row stride 520 halfs ----
  const unsigned long long gaddr = (unsigned long long)(uintptr_t)hbuf;
  const unsigned ldsAddr = (unsigned)(uintptr_t)(void*)sAh;   // LDS byte offset
  v4u g0;
  g0.x = 1u;                                        // count=1 valid descriptor
  g0.y = ldsAddr;                                   // lds_addr
  g0.z = (unsigned)(gaddr & 0xFFFFFFFFull);         // global_addr[31:0]
  g0.w = (unsigned)((gaddr >> 32) & 0x01FFFFFFull)  // global_addr[56:32]
         | (2u << 30);                              // type = 2 (image)
  v8i g1;
  g1[0] = (int)((1u << 16)      // data_size = 2 bytes (bf16)
              | (1u << 20)      // pad_enable
              | (7u << 22)      // pad_interval: 256 DWORDs
              | (3u << 25));    // pad_amount : 4 DWORDs
  g1[1] = (int)(512u << 16);    // tensor_dim0[15:0]=512 in bits[63:48]
  g1[2] = (int)(32u  << 16);    // tensor_dim1[15:0]=32  in bits[95:80]
  g1[3] = (int)(512u << 16);    // tile_dim0 = 512       in bits[127:112]
  g1[4] = 32;                   // tile_dim1 = 32        in bits[143:128]
  g1[5] = 512;                  // tensor_dim0_stride = 512 (low 32 bits)
  g1[6] = 0;
  g1[7] = 0;
  const v4i gz = {0, 0, 0, 0};

  // per-thread recurrent state: pairs p = tid, tid+256 ; p = b*16 + jj
  const int b0 = tid >> 4,        jj0 = tid & 15;
  const int b1 = (tid >> 4) + 16, jj1 = tid & 15;
  float c0 = 0.f, h0 = 0.f, c1 = 0.f, h1 = 0.f;

  // WMMA tile assignment: wave -> (m0, n0)
  const int wave = tid >> 5, lane = tid & 31;
  const int m0 = (wave >> 2) << 4;                 // 0 or 16 (batch rows)
  const int n0 = (wave & 3) << 4;                  // 0/16/32/48 (gate cols)
  // CDNA5 16-bit A 16x32 layout: lanes 0-15 -> K {0..7,16..23}, lanes 16-31 -> K {8..15,24..31}
  const int arow  = m0 + (lane & 15);
  const int kbase = (lane & 16) ? 8 : 0;
  // B 32x16 layout: lane&15 = column N, lanes 0-15 -> K {0..7,16..23}, lanes 16-31 -> K {8..15,24..31}
  const int brow = n0 + (lane & 15);
  const int bko  = (lane & 16) ? 8 : 0;
  const int gcol = n0 + (lane & 15);

  for (int step = 0; step < TT; ++step){
    const int t = dir ? (TT - 1 - step) : step;

    // ---- TDM: async h (bf16, L2-resident) -> sAh ; wave 0 issues the DMA ----
    if (wave == 0){
#if __clang_major__ >= 23
      __builtin_amdgcn_tensor_load_to_lds(g0, g1, gz, gz, g1, 0);
#else
      __builtin_amdgcn_tensor_load_to_lds(g0, g1, gz, gz, 0);
#endif
    }

    // ---- stage x_t (fp32 -> bf16) into sAx, prefetch next timestep's x ----
    {
      const int tn = dir ? (t > 0 ? t - 1 : t) : (t < TT - 1 ? t + 1 : t);
      for (int idx = tid; idx < BB * (DD / 4); idx += 256){
        int b = idx >> 7, k4 = (idx & 127) << 2;
        const float* src = &x[((size_t)(b * TT + t)) * DD + k4];
        __builtin_prefetch(&x[((size_t)(b * TT + tn)) * DD + k4], 0, 0);
        const float4 v = *(const float4*)src;
        unsigned short* p = &sAx[b * AXSTR + k4];
        p[0] = f2bf(v.x); p[1] = f2bf(v.y); p[2] = f2bf(v.z); p[3] = f2bf(v.w);
      }
    }
    if (wave == 0) __builtin_amdgcn_s_wait_tensorcnt(0);
    __syncthreads();

    // ---- fused GEMM: gates[32 x 64] = [x_t | h] (32x1024) x W^T (1024x64) ----
    v8f acc = {0.f,0.f,0.f,0.f,0.f,0.f,0.f,0.f};
    #pragma unroll
    for (int half = 0; half < 2; ++half){
      const unsigned short* aBase = half ? &sAh[arow * AHSTR + kbase]
                                         : &sAx[arow * AXSTR + kbase];
      const unsigned short* bBase = &sW[brow * WSTR + half * DD + bko];
      #pragma unroll 4
      for (int kc = 0; kc < 16; ++kc){
        bf16x8 alo = *(const bf16x8*)(aBase + kc * 32);
        bf16x8 ahi = *(const bf16x8*)(aBase + kc * 32 + 16);
        v16bf av = __builtin_shufflevector(alo, ahi,
                     0,1,2,3,4,5,6,7,8,9,10,11,12,13,14,15);
        bf16x8 blo = *(const bf16x8*)(bBase + kc * 32);
        bf16x8 bhi = *(const bf16x8*)(bBase + kc * 32 + 16);
        v16bf bv = __builtin_shufflevector(blo, bhi,
                     0,1,2,3,4,5,6,7,8,9,10,11,12,13,14,15);
        acc = __builtin_amdgcn_wmma_f32_16x16x32_bf16(
                  false, av, false, bv, (short)0, acc, false, false);
      }
    }

    // scatter gate tile (+bias) to LDS; C layout: VGPR v -> M = m0+v (+8 for lanes>=16)
    {
      float bia  = sBias[gcol];
      int rbase = m0 + ((lane & 16) ? 8 : 0);
      #pragma unroll
      for (int v = 0; v < 8; ++v)
        sG[(rbase + v) * GSTR + gcol] = acc[v] + bia;
    }
    __syncthreads();

    // ---- LSTM cell: 2 (b, j) pairs per thread, state in registers ----
    {
      float ig = sigmoidf_(sG[b0 * GSTR +      jj0]);
      float fg = sigmoidf_(sG[b0 * GSTR + 16 + jj0]);
      float gg = tanhf_   (sG[b0 * GSTR + 32 + jj0]);
      float og = sigmoidf_(sG[b0 * GSTR + 48 + jj0]);
      float cn = fg * c0 + ig * gg;
      float hn = og * tanhf_(cn);
      if (t < sLen[b0]) { c0 = cn; h0 = hn; }
      hbuf[b0 * HH + j0 + jj0] = f2bf(h0);
      out[((size_t)(b0 * TT + t)) * (2 * HH) + dir * HH + j0 + jj0] = h0;
    }
    {
      float ig = sigmoidf_(sG[b1 * GSTR +      jj1]);
      float fg = sigmoidf_(sG[b1 * GSTR + 16 + jj1]);
      float gg = tanhf_   (sG[b1 * GSTR + 32 + jj1]);
      float og = sigmoidf_(sG[b1 * GSTR + 48 + jj1]);
      float cn = fg * c1 + ig * gg;
      float hn = og * tanhf_(cn);
      if (t < sLen[b1]) { c1 = cn; h1 = hn; }
      hbuf[b1 * HH + j0 + jj1] = f2bf(h1);
      out[((size_t)(b1 * TT + t)) * (2 * HH) + dir * HH + j0 + jj1] = h1;
    }

    // ---- device-wide step barrier (per direction, monotonic counter) ----
    __builtin_amdgcn_fence(__ATOMIC_RELEASE, "agent");
    __syncthreads();
    if (tid == 0){
      __hip_atomic_fetch_add(barp, 1u, __ATOMIC_RELAXED, __HIP_MEMORY_SCOPE_AGENT);
      unsigned tgt = (unsigned)(step + 1) * (unsigned)WGD;
      while (__hip_atomic_load(barp, __ATOMIC_RELAXED, __HIP_MEMORY_SCOPE_AGENT) < tgt)
        __builtin_amdgcn_s_sleep(8);
    }
    __syncthreads();
    __builtin_amdgcn_fence(__ATOMIC_ACQUIRE, "agent");
  }
}

extern "C" void kernel_launch(void* const* d_in, const int* in_sizes, int n_in,
                              void* d_out, int out_size, void* d_ws, size_t ws_size,
                              hipStream_t stream){
  (void)in_sizes; (void)n_in; (void)out_size; (void)ws_size;
  const float* x     = (const float*)d_in[0];
  const int*   len   = (const int*)  d_in[1];
  const float* Wih_f = (const float*)d_in[2];
  const float* Whh_f = (const float*)d_in[3];
  const float* bih_f = (const float*)d_in[4];
  const float* bhh_f = (const float*)d_in[5];
  const float* Wih_b = (const float*)d_in[6];
  const float* Whh_b = (const float*)d_in[7];
  const float* bih_b = (const float*)d_in[8];
  const float* bhh_b = (const float*)d_in[9];
  float* out = (float*)d_out;

  unsigned char* ws = (unsigned char*)d_ws;
  unsigned*       bar = (unsigned*)ws;                          // 2 counters @ 0
  unsigned short* h_f = (unsigned short*)(ws + 256);            // 32 KB
  unsigned short* h_b = (unsigned short*)(ws + 256 + BB*HH*2);  // 32 KB

  int nwords = (256 + 2 * BB * HH * 2) / 4;
  bilstm_init<<<dim3((nwords + 255) / 256), dim3(256), 0, stream>>>(
      (unsigned*)ws, nwords);

  bilstm_persistent<<<dim3(2 * WGD), dim3(256), 0, stream>>>(
      x, len, Wih_f, Whh_f, bih_f, bhh_f, Wih_b, Whh_b, bih_b, bhh_b,
      out, h_f, h_b, bar);
}
